// MOLT_56968446214731
// MI455X (gfx1250) — matrix operand order repaired and checked by reading, not compiled, not run on hardware
//
#include <hip/hip_runtime.h>
#include <hip/hip_bf16.h>

// ---------------------------------------------------------------------------
// MOLT forward for MI455X (gfx1250, wave32, WMMA bf16 16x16x32).
// Two-stage fused GEMM chain per 16-row batch tile; V/U pre-converted to bf16
// in workspace (fallback to direct f32 reads if workspace too small).
// ---------------------------------------------------------------------------

typedef __attribute__((ext_vector_type(16))) __bf16 v16bf;
typedef __attribute__((ext_vector_type(8)))  float  v8f;

union F16x { v16bf v; __bf16 e[16]; unsigned int u[8]; };
union F8x  { v8f   v; float  e[8]; };

#define D_MODEL   768
#define BATCH     4096
#define KG        2048            // stacked n*r per group (same for all 5 groups)
#define BT        16              // batch rows per workgroup
#define NTOT      124             // total transforms
#define XS_STRIDE 776             // bf16 elems per x-tile row (768 + pad)
#define PS_STRIDE 136             // bf16 elems per Pg row (128 + pad)

#define OUT_ELEMS (BATCH * D_MODEL)   // 3145728
#define SP_OFF    (OUT_ELEMS)
#define ACT_OFF   (OUT_ELEMS + 1)

#define GRP_ELEMS ((size_t)KG * D_MODEL)        // 1572864 per group (V or U)
#define WS_BF_OFF 4096                          // byte offset of bf16 region in ws

__device__ __forceinline__ void map_n(int n, int& g, int& nloc, int& r) {
    if      (n < 4)  { g = 0; nloc = n;      r = 512; }
    else if (n < 12) { g = 1; nloc = n - 4;  r = 256; }
    else if (n < 28) { g = 2; nloc = n - 12; r = 128; }
    else if (n < 60) { g = 3; nloc = n - 28; r = 64;  }
    else             { g = 4; nloc = n - 60; r = 32;  }
}

// ---------------------------------------------------------------------------
// scratch zeroing (graph-capture safe)
// ws[0..123] : sum_b |gate| ; ws[124] : active count
// ws[128+2n] : ||V_n||^2    ; ws[129+2n] : ||U_n||^2
// ---------------------------------------------------------------------------
__global__ void molt_zero_ws(float* ws) {
    if (threadIdx.x < 512) ws[threadIdx.x] = 0.0f;
}

// ---------------------------------------------------------------------------
// one-shot f32 -> bf16 conversion of stacked V and U into workspace
// ---------------------------------------------------------------------------
__global__ __launch_bounds__(256) void molt_convert(
    const float* V0, const float* V1, const float* V2, const float* V3, const float* V4,
    const float* U0, const float* U1, const float* U2, const float* U3, const float* U4,
    __bf16* __restrict__ Vbf, __bf16* __restrict__ Ubf) {
    const float* Vs[5] = {V0, V1, V2, V3, V4};
    const float* Us[5] = {U0, U1, U2, U3, U4};
    const size_t per4 = GRP_ELEMS / 4;
    const size_t tot4 = 5 * per4;
    for (size_t i = (size_t)blockIdx.x * 256 + threadIdx.x; i < tot4;
         i += (size_t)gridDim.x * 256) {
        size_t g = i / per4, rem = i % per4;
        float4 v = ((const float4*)Vs[g])[rem];
        float4 u = ((const float4*)Us[g])[rem];
        __bf16* dv = Vbf + g * GRP_ELEMS + rem * 4;
        __bf16* du = Ubf + g * GRP_ELEMS + rem * 4;
        dv[0] = (__bf16)v.x; dv[1] = (__bf16)v.y; dv[2] = (__bf16)v.z; dv[3] = (__bf16)v.w;
        du[0] = (__bf16)u.x; du[1] = (__bf16)u.y; du[2] = (__bf16)u.z; du[3] = (__bf16)u.w;
    }
}

// ---------------------------------------------------------------------------
// per-transform Frobenius sums-of-squares (124 blocks x 256 threads)
// ---------------------------------------------------------------------------
__global__ __launch_bounds__(256) void molt_norms(
    const float* V0, const float* V1, const float* V2, const float* V3, const float* V4,
    const float* U0, const float* U1, const float* U2, const float* U3, const float* U4,
    float* ws) {
    const float* Vs[5] = {V0, V1, V2, V3, V4};
    const float* Us[5] = {U0, U1, U2, U3, U4};
    int n = blockIdx.x, g, nloc, r;
    map_n(n, g, nloc, r);
    size_t cnt = (size_t)r * D_MODEL;
    const float* vp = Vs[g] + (size_t)nloc * cnt;
    const float* up = Us[g] + (size_t)nloc * cnt;
    float sv = 0.0f, su = 0.0f;
    for (size_t i = threadIdx.x; i < cnt; i += 256) {
        float a = vp[i]; sv += a * a;
        float b = up[i]; su += b * b;
    }
    __shared__ float r1[256], r2[256];
    r1[threadIdx.x] = sv; r2[threadIdx.x] = su;
    __syncthreads();
    for (int s = 128; s > 0; s >>= 1) {
        if ((int)threadIdx.x < s) {
            r1[threadIdx.x] += r1[threadIdx.x + s];
            r2[threadIdx.x] += r2[threadIdx.x + s];
        }
        __syncthreads();
    }
    if (threadIdx.x == 0) { ws[128 + 2 * n] = r1[0]; ws[129 + 2 * n] = r2[0]; }
}

// ---------------------------------------------------------------------------
// main fused kernel: gates + (x Vt) * gate * Ut per 16-row batch tile
// PRE=true : read pre-converted bf16 V/U from workspace (pure b128 + wmma loop)
// PRE=false: read f32 V/U and convert in-loop (fallback)
// ---------------------------------------------------------------------------
template <bool PRE>
__global__ __launch_bounds__(256) void molt_main(
    const float* __restrict__ x,
    const float* V0, const float* V1, const float* V2, const float* V3, const float* V4,
    const float* U0, const float* U1, const float* U2, const float* U3, const float* U4,
    const float* e0, const float* e1, const float* e2, const float* e3, const float* e4,
    const float* b0, const float* b1, const float* b2, const float* b3, const float* b4,
    const __bf16* __restrict__ Vbf, const __bf16* __restrict__ Ubf,
    float* __restrict__ out, float* __restrict__ ws) {

    __shared__ __bf16 xs[BT * XS_STRIDE];   // x tile, bf16
    __shared__ float  gs[BT * 128];         // gates for this tile (stride 128)
    __shared__ __bf16 pgt[BT * PS_STRIDE];  // gated P^T chunk, bf16
    __shared__ float  red[256];

    const float* Vs[5]   = {V0, V1, V2, V3, V4};
    const float* Us[5]   = {U0, U1, U2, U3, U4};
    const float* encs[5] = {e0, e1, e2, e3, e4};
    const float* bias[5] = {b0, b1, b2, b3, b4};
    const int rr[5]   = {512, 256, 128, 64, 32};
    const int rsft[5] = {9, 8, 7, 6, 5};
    const int goff[5] = {0, 4, 12, 28, 60};
    const int ng[5]   = {4, 8, 16, 32, 64};
    const long gate_off[5] = {
        (long)OUT_ELEMS + 2,
        (long)OUT_ELEMS + 2 + 4096L * 4,
        (long)OUT_ELEMS + 2 + 4096L * 12,
        (long)OUT_ELEMS + 2 + 4096L * 28,
        (long)OUT_ELEMS + 2 + 4096L * 60,
    };

    const int tid  = threadIdx.x;
    const int wave = tid >> 5;
    const int lane = tid & 31;
    const int l15  = lane & 15;
    const bool hi  = lane >= 16;
    const int koff8  = hi ? 8 : 0;
    const int koff16 = hi ? 16 : 0;
    const int bt0  = blockIdx.x * BT;

    // ---- stage x tile into LDS as bf16 ------------------------------------
    for (int i = tid; i < BT * (D_MODEL / 4); i += 256) {
        int row = i / (D_MODEL / 4), c4 = i % (D_MODEL / 4);
        float4 v = ((const float4*)x)[(size_t)(bt0 + row) * (D_MODEL / 4) + c4];
        __bf16* d = &xs[row * XS_STRIDE + c4 * 4];
        d[0] = (__bf16)v.x; d[1] = (__bf16)v.y; d[2] = (__bf16)v.z; d[3] = (__bf16)v.w;
    }
    __syncthreads();

    // ---- gates: pre = x.enc - b ; jumprelu --------------------------------
    for (int idx = tid; idx < BT * NTOT; idx += 256) {
        int b = idx / NTOT, n = idx % NTOT;
        int g, nloc, r;
        map_n(n, g, nloc, r);
        const float*  ep = encs[g] + (size_t)nloc * D_MODEL;
        const __bf16* xr = &xs[b * XS_STRIDE];
        float a = 0.0f;
        #pragma unroll 4
        for (int k = 0; k < D_MODEL; ++k) a += (float)xr[k] * ep[k];
        float pre  = a - bias[g][nloc];
        float gate = pre > 0.0f ? pre : 0.0f;
        gs[b * 128 + n] = gate;
        out[gate_off[g] + (size_t)(bt0 + b) * ng[g] + nloc] = gate;
    }
    __syncthreads();

    // ---- tile-level gate reductions ---------------------------------------
    float cnt = 0.0f;
    if (tid < NTOT) {
        float s = 0.0f; int c = 0;
        #pragma unroll
        for (int b = 0; b < BT; ++b) {
            float v = gs[b * 128 + tid];
            s += fabsf(v);
            c += (v > 0.0f) ? 1 : 0;
        }
        atomicAdd(&ws[tid], s);
        cnt = (float)c;
    }
    red[tid] = cnt;
    __syncthreads();
    for (int s = 128; s > 0; s >>= 1) {
        if (tid < s) red[tid] += red[tid + s];
        __syncthreads();
    }
    if (tid == 0) atomicAdd(&ws[124], red[0]);

    // ---- persistent f32 accumulators: wave owns d-cols [96w, 96w+96) ------
    v8f acc[6];
    #pragma unroll
    for (int t = 0; t < 6; ++t) acc[t] = (v8f){0.f, 0.f, 0.f, 0.f, 0.f, 0.f, 0.f, 0.f};

    for (int g = 0; g < 5; ++g) {
        const float*  Vg  = Vs[g];
        const float*  Ug  = Us[g];
        const __bf16* Vbg = Vbf + (size_t)g * GRP_ELEMS;
        const __bf16* Ubg = Ubf + (size_t)g * GRP_ELEMS;
        const int r = rr[g], sft = rsft[g], go = goff[g];

        for (int kc = 0; kc < KG; kc += 128) {
            // ===== GEMM1: P^T(128x16) = Vstack_chunk(128x768) . x_tile^T ====
            v8f p = (v8f){0.f, 0.f, 0.f, 0.f, 0.f, 0.f, 0.f, 0.f};
            const int vrow = kc + 16 * wave + l15;
            const float*  arow  = Vg  + (size_t)vrow * D_MODEL;
            const __bf16* abrow = Vbg + (size_t)vrow * D_MODEL;
            if (kc + 128 < KG) {
                if constexpr (PRE)
                    __builtin_prefetch(Vbg + (size_t)(vrow + 128) * D_MODEL, 0, 1);
                else
                    __builtin_prefetch(Vg + (size_t)(vrow + 128) * D_MODEL, 0, 1);
            }

            #pragma unroll 4
            for (int k0 = 0; k0 < D_MODEL; k0 += 32) {
                // A fragment: V row, K in {k0..+7, k0+16..+23} (+8 for hi lanes)
                F16x A;
                if constexpr (PRE) {
                    uint4 qa = ((const uint4*)(abrow + k0 + koff8))[0];
                    uint4 qb = ((const uint4*)(abrow + k0 + 16 + koff8))[0];
                    A.u[0] = qa.x; A.u[1] = qa.y; A.u[2] = qa.z; A.u[3] = qa.w;
                    A.u[4] = qb.x; A.u[5] = qb.y; A.u[6] = qb.z; A.u[7] = qb.w;
                } else {
                    const float4* f1 = (const float4*)(arow + k0 + koff8);
                    const float4* f2 = (const float4*)(arow + k0 + 16 + koff8);
                    float4 a0 = f1[0], a1 = f1[1], a2 = f2[0], a3 = f2[1];
                    A.e[0]  = (__bf16)a0.x; A.e[1]  = (__bf16)a0.y; A.e[2]  = (__bf16)a0.z; A.e[3]  = (__bf16)a0.w;
                    A.e[4]  = (__bf16)a1.x; A.e[5]  = (__bf16)a1.y; A.e[6]  = (__bf16)a1.z; A.e[7]  = (__bf16)a1.w;
                    A.e[8]  = (__bf16)a2.x; A.e[9]  = (__bf16)a2.y; A.e[10] = (__bf16)a2.z; A.e[11] = (__bf16)a2.w;
                    A.e[12] = (__bf16)a3.x; A.e[13] = (__bf16)a3.y; A.e[14] = (__bf16)a3.z; A.e[15] = (__bf16)a3.w;
                }
                // B fragment: x row l15, K in {k0..+7, k0+16..+23} (+8 for hi lanes)
                F16x B;
                {
                    const uint4* bp  = (const uint4*)&xs[l15 * XS_STRIDE + k0 + koff8];
                    const uint4* bp2 = (const uint4*)&xs[l15 * XS_STRIDE + k0 + 16 + koff8];
                    uint4 q0 = bp[0], q1 = bp2[0];
                    B.u[0] = q0.x; B.u[1] = q0.y; B.u[2] = q0.z; B.u[3] = q0.w;
                    B.u[4] = q1.x; B.u[5] = q1.y; B.u[6] = q1.z; B.u[7] = q1.w;
                }
                p = __builtin_amdgcn_wmma_f32_16x16x32_bf16(
                        false, A.v, false, B.v, (short)0, p, false, false);
            }

            // ===== gate (jumprelu) + store P^T chunk to LDS as bf16 =========
            {
                F8x pp; pp.v = p;
                #pragma unroll
                for (int q = 0; q < 8; ++q) {
                    int m = q + (hi ? 8 : 0);
                    int c = kc + 16 * wave + m;          // stacked column in group
                    int j = c >> sft;                    // transform index
                    float gate = gs[l15 * 128 + go + j];
                    pgt[l15 * PS_STRIDE + 16 * wave + m] = (__bf16)(pp.e[q] * gate);
                }
            }
            __syncthreads();

            // ===== GEMM2: out_tile += Pg^T(16x128) . Ustack_chunk(128x768) ==
            #pragma unroll
            for (int k0 = 0; k0 < 128; k0 += 32) {
                F16x A2;
                {
                    const uint4* pa = (const uint4*)&pgt[l15 * PS_STRIDE + k0 + koff16];
                    const uint4* pb = (const uint4*)&pgt[l15 * PS_STRIDE + k0 + 8 + koff16];
                    uint4 qa = pa[0], qb = pb[0];
                    A2.u[0] = qa.x; A2.u[1] = qa.y; A2.u[2] = qa.z; A2.u[3] = qa.w;
                    A2.u[4] = qb.x; A2.u[5] = qb.y; A2.u[6] = qb.z; A2.u[7] = qb.w;
                }
                int kglob = kc + k0;
                int j2 = kglob >> sft;                  // 32-slice stays in one transform
                int kl = kglob & (r - 1);
                #pragma unroll
                for (int t = 0; t < 6; ++t) {
                    int dd = wave * 96 + t * 16 + l15;
                    F16x B2;
                    if constexpr (PRE) {
                        const __bf16* ub = Ubg + (size_t)j2 * D_MODEL * r + kl + koff16
                                               + (size_t)dd * r;
                        uint4 q0 = ((const uint4*)ub)[0];
                        uint4 q1 = ((const uint4*)ub)[1];
                        B2.u[0] = q0.x; B2.u[1] = q0.y; B2.u[2] = q0.z; B2.u[3] = q0.w;
                        B2.u[4] = q1.x; B2.u[5] = q1.y; B2.u[6] = q1.z; B2.u[7] = q1.w;
                    } else {
                        const float* uf = Ug + (size_t)j2 * D_MODEL * r + kl + koff16
                                             + (size_t)dd * r;
                        float4 u0 = ((const float4*)uf)[0];
                        float4 u1 = ((const float4*)uf)[1];
                        float4 u2 = ((const float4*)uf)[2];
                        float4 u3 = ((const float4*)uf)[3];
                        B2.e[0]  = (__bf16)u0.x; B2.e[1]  = (__bf16)u0.y; B2.e[2]  = (__bf16)u0.z; B2.e[3]  = (__bf16)u0.w;
                        B2.e[4]  = (__bf16)u1.x; B2.e[5]  = (__bf16)u1.y; B2.e[6]  = (__bf16)u1.z; B2.e[7]  = (__bf16)u1.w;
                        B2.e[8]  = (__bf16)u2.x; B2.e[9]  = (__bf16)u2.y; B2.e[10] = (__bf16)u2.z; B2.e[11] = (__bf16)u2.w;
                        B2.e[12] = (__bf16)u3.x; B2.e[13] = (__bf16)u3.y; B2.e[14] = (__bf16)u3.z; B2.e[15] = (__bf16)u3.w;
                    }
                    acc[t] = __builtin_amdgcn_wmma_f32_16x16x32_bf16(
                                 false, A2.v, false, B2.v, (short)0, acc[t], false, false);
                }
            }
            __syncthreads();
        }
    }

    // ---- write out tile ---------------------------------------------------
    #pragma unroll
    for (int t = 0; t < 6; ++t) {
        F8x aa; aa.v = acc[t];
        int col = wave * 96 + t * 16 + l15;
        #pragma unroll
        for (int q = 0; q < 8; ++q) {
            int m = q + (hi ? 8 : 0);
            out[(size_t)(bt0 + m) * D_MODEL + col] = aa.e[q];
        }
    }
}

// ---------------------------------------------------------------------------
// finalize scalars
// ---------------------------------------------------------------------------
__global__ void molt_finalize(const float* ws, float* out) {
    __shared__ float red[128];
    int t = threadIdx.x;
    float v = 0.0f;
    if (t < NTOT) {
        int g, nloc, r;
        map_n(t, g, nloc, r);
        float mean = ws[t] * (1.0f / (float)BATCH);
        float frob = sqrtf(ws[128 + 2 * t] * ws[129 + 2 * t]) /
                     sqrtf((float)(D_MODEL * r));
        v = tanhf(mean) * frob;
    }
    red[t] = v;
    __syncthreads();
    for (int s = 64; s > 0; s >>= 1) {
        if (t < s) red[t] += red[t + s];
        __syncthreads();
    }
    if (t == 0) {
        out[SP_OFF]  = red[0];
        out[ACT_OFF] = ws[124] * (1.0f / (float)BATCH);
    }
}

// ---------------------------------------------------------------------------
extern "C" void kernel_launch(void* const* d_in, const int* in_sizes, int n_in,
                              void* d_out, int out_size, void* d_ws, size_t ws_size,
                              hipStream_t stream) {
    const float* x  = (const float*)d_in[0];
    const float* V[5]; const float* U[5]; const float* E[5]; const float* B[5];
    for (int i = 0; i < 5; ++i) {
        V[i] = (const float*)d_in[1 + 4 * i];
        U[i] = (const float*)d_in[2 + 4 * i];
        E[i] = (const float*)d_in[3 + 4 * i];
        B[i] = (const float*)d_in[4 + 4 * i];
    }
    float* out = (float*)d_out;
    float* ws  = (float*)d_ws;

    const size_t need = WS_BF_OFF + 2 * (5 * GRP_ELEMS) * sizeof(__bf16); // ~31.5 MB
    const bool   pre  = ws_size >= need;
    __bf16* Vbf = (__bf16*)((char*)d_ws + WS_BF_OFF);
    __bf16* Ubf = Vbf + 5 * GRP_ELEMS;

    molt_zero_ws<<<1, 512, 0, stream>>>(ws);
    if (pre) {
        molt_convert<<<2048, 256, 0, stream>>>(V[0], V[1], V[2], V[3], V[4],
                                               U[0], U[1], U[2], U[3], U[4],
                                               Vbf, Ubf);
    }
    molt_norms<<<124, 256, 0, stream>>>(V[0], V[1], V[2], V[3], V[4],
                                        U[0], U[1], U[2], U[3], U[4], ws);
    if (pre) {
        molt_main<true><<<BATCH / BT, 256, 0, stream>>>(
            x, V[0], V[1], V[2], V[3], V[4], U[0], U[1], U[2], U[3], U[4],
            E[0], E[1], E[2], E[3], E[4], B[0], B[1], B[2], B[3], B[4],
            Vbf, Ubf, out, ws);
    } else {
        molt_main<false><<<BATCH / BT, 256, 0, stream>>>(
            x, V[0], V[1], V[2], V[3], V[4], U[0], U[1], U[2], U[3], U[4],
            E[0], E[1], E[2], E[3], E[4], B[0], B[1], B[2], B[3], B[4],
            Vbf, Ubf, out, ws);
    }
    molt_finalize<<<1, 128, 0, stream>>>(ws, out);
}
